// RecurrentLayer_24103356465315
// MI455X (gfx1250) — matrix operand order repaired and checked by reading, not compiled, not run on hardware
//
#include <hip/hip_runtime.h>
#include <hip/hip_bf16.h>
#include <math.h>

#define SEQ      512
#define BATCH    64
#define IN_DIM   128
#define HID      512
#define NLAYERS  3
#define ALPHA    1.0f
#define PRE_NS   0.1f
#define POST_NS  0.1f
#define NBLK_RNN 16   // one cluster of 16 workgroups, 32 hidden cols each

typedef __attribute__((ext_vector_type(16))) __bf16        v16bf;
typedef __attribute__((ext_vector_type(8)))  float         v8f;
typedef __attribute__((ext_vector_type(4)))  unsigned int  u32x4;

union Frag {
    v16bf          v;
    u32x4          q[2];
    unsigned short s[16];
};

// ---- float -> bf16 (round to nearest even) --------------------------------
__device__ __forceinline__ unsigned short f2bf(float f) {
    unsigned u = __float_as_uint(f);
    unsigned r = u + 0x7FFFu + ((u >> 16) & 1u);
    return (unsigned short)(r >> 16);
}

// ---- JAX threefry2x32 (20 rounds) -----------------------------------------
struct U2 { unsigned x, y; };

__device__ __forceinline__ unsigned rotl32(unsigned x, int n) {
    return (x << n) | (x >> (32 - n));
}

__device__ __forceinline__ U2 threefry2x32(U2 key, U2 ctr) {
    const unsigned ks0 = key.x, ks1 = key.y, ks2 = ks0 ^ ks1 ^ 0x1BD11BDAu;
    unsigned x0 = ctr.x + ks0, x1 = ctr.y + ks1;
    const int rot[8] = {13, 15, 26, 6, 17, 29, 16, 24};
#pragma unroll
    for (int g = 0; g < 5; ++g) {
#pragma unroll
        for (int r = 0; r < 4; ++r) {
            x0 += x1;
            x1 = rotl32(x1, rot[(g & 1) * 4 + r]);
            x1 ^= x0;
        }
        unsigned i0, i1;
        switch (g) {               // subkey injection schedule d = 1..5
            case 0:  i0 = ks1; i1 = ks2 + 1u; break;
            case 1:  i0 = ks2; i1 = ks0 + 2u; break;
            case 2:  i0 = ks0; i1 = ks1 + 3u; break;
            case 3:  i0 = ks1; i1 = ks2 + 4u; break;
            default: i0 = ks2; i1 = ks0 + 5u; break;
        }
        x0 += i0; x1 += i1;
    }
    U2 o; o.x = x0; o.y = x1; return o;
}

__device__ __forceinline__ U2 jax_fold_in(U2 key, unsigned d) {
    U2 c; c.x = 0u; c.y = d;
    return threefry2x32(key, c);
}

// element idx of a (BATCH*HID)-sample normal draw, JAX split-counter layout
__device__ __forceinline__ float jax_normal(U2 key, unsigned idx) {
    const unsigned half = (BATCH * HID) >> 1;
    U2 ctr;
    if (idx < half) { ctr.x = idx;        ctr.y = idx + half; }
    else            { ctr.x = idx - half; ctr.y = idx;        }
    U2 r = threefry2x32(key, ctr);
    unsigned bits = (idx < half) ? r.x : r.y;
    float u = __uint_as_float((bits >> 9) | 0x3F800000u) - 1.0f;  // [0,1)
    u = 2.0f * u - 1.0f;                                          // [-1,1)
    u = fmaxf(u, -0.999999940395355225f);                         // avoid erfinv(-1)
    return 1.41421356237309515f * erfinvf(u);
}

// ---- WMMA wrapper ----------------------------------------------------------
__device__ __forceinline__ v8f wmma_bf16(const Frag& a, const Frag& b, v8f c) {
    return __builtin_amdgcn_wmma_f32_16x16x32_bf16(
        false, a.v, false, b.v, (short)0, c, false, false);
}

// ---- cross-WGP barrier: CDNA5 cluster barrier + atomic fallback ------------
__device__ unsigned g_bar_cnt = 0;
__device__ unsigned g_bar_gen = 0;

__device__ __forceinline__ void cluster_sync() {
    __syncthreads();
    __builtin_amdgcn_fence(__ATOMIC_RELEASE, "agent");
    // s_barrier_signal -3 ; s_barrier_wait -3  (NOP if not cluster-dispatched)
    __builtin_amdgcn_s_cluster_barrier();
    if (threadIdx.x == 0) {
        unsigned gen = __hip_atomic_load(&g_bar_gen, __ATOMIC_RELAXED,
                                         __HIP_MEMORY_SCOPE_AGENT);
        unsigned arrived = __hip_atomic_fetch_add(&g_bar_cnt, 1u,
                                                  __ATOMIC_ACQ_REL,
                                                  __HIP_MEMORY_SCOPE_AGENT);
        if (arrived == (unsigned)(NBLK_RNN - 1)) {
            __hip_atomic_store(&g_bar_cnt, 0u, __ATOMIC_RELAXED,
                               __HIP_MEMORY_SCOPE_AGENT);
            __hip_atomic_fetch_add(&g_bar_gen, 1u, __ATOMIC_RELEASE,
                                   __HIP_MEMORY_SCOPE_AGENT);
        } else {
            while (__hip_atomic_load(&g_bar_gen, __ATOMIC_ACQUIRE,
                                     __HIP_MEMORY_SCOPE_AGENT) == gen) {
                __builtin_amdgcn_s_sleep(1);
            }
        }
    }
    __syncthreads();
    __builtin_amdgcn_fence(__ATOMIC_ACQUIRE, "agent");
}

// ===========================================================================
// Kernel 1: WU[s,b,h] = inp[s,b,:] . W_in[h,:] + b_in[h]   (WMMA bf16 GEMM)
// ===========================================================================
__global__ __launch_bounds__(256) void lifrnn_inproj(
    const float* __restrict__ inp, const float* __restrict__ W_in,
    const float* __restrict__ b_in, float* __restrict__ WU) {
    const int lane = threadIdx.x & 31, wid = threadIdx.x >> 5;
    const int tile = blockIdx.x * 8 + wid;   // (SEQ*BATCH/16) * (HID/16) tiles
    const int mt = tile >> 5;                // row-tile over M = SEQ*BATCH
    const int nt = tile & 31;                // col-tile over N = HID
    const int half = lane >> 4, l16 = lane & 15;
    const int mrow = mt * 16 + l16;          // A row for this lane
    const int ncol = nt * 16 + l16;          // B col for this lane
    v8f c = {};
#pragma unroll
    for (int kb = 0; kb < IN_DIM / 32; ++kb) {
        Frag a, b;
#pragma unroll
        for (int e = 0; e < 16; ++e) {       // ISA 16-bit A layout
            int k = kb * 32 + ((e < 8) ? (e + 8 * half) : (e + 8 + 8 * half));
            a.s[e] = f2bf(inp[mrow * IN_DIM + k]);
        }
        const float* wrow = W_in + ncol * IN_DIM + kb * 32;
#pragma unroll
        for (int e = 0; e < 16; ++e)         // ISA 16-bit B layout (same K order as A)
            b.s[e] = f2bf(wrow[(e < 8) ? (e + 8 * half) : (e + 8 + 8 * half)]);
        c = wmma_bf16(a, b, c);
    }
    const float bias = b_in[nt * 16 + l16];
#pragma unroll
    for (int r = 0; r < 8; ++r) {            // C layout: M = r + 8*half, N = l16
        int row = mt * 16 + r + 8 * half;
        __builtin_nontemporal_store(c[r] + bias,
                                    &WU[row * HID + nt * 16 + l16]);
    }
}

// ===========================================================================
// Kernel 2: persistent 16-WGP recurrent scan. Weight slice lives in LDS,
// fr activation ping-pongs through L2, v-state lives in registers.
// ===========================================================================
__global__ __launch_bounds__(256) void lifrnn_scan(
    const float* __restrict__ W_h, const float* __restrict__ b_h,
    const float* __restrict__ WU, unsigned short* __restrict__ frbuf,
    float* __restrict__ out) {
    extern __shared__ unsigned short sWh[];  // [NLAYERS][32][HID] bf16 = 96 KB
    const int hbase = blockIdx.x * 32;

    // Stage this WGP's weight slice (transposed: [j][n][k], k contiguous)
    for (int i = threadIdx.x; i < NLAYERS * 32 * HID; i += blockDim.x) {
        int j = i / (32 * HID);
        int rem = i - j * 32 * HID;
        int n = rem / HID, k = rem - n * HID;
        sWh[i] = f2bf(W_h[(j * HID + hbase + n) * HID + k]);
    }
    // fr0 = relu(0) = 0 in ping buffer 0 (our slice)
    for (int i = threadIdx.x; i < BATCH * 32; i += blockDim.x) {
        int b = i >> 5, n = i & 31;
        frbuf[b * HID + hbase + n] = 0;
    }
    cluster_sync();

    const int lane = threadIdx.x & 31, wid = threadIdx.x >> 5;
    const int half = lane >> 4, l16 = lane & 15;
    const int mt = wid >> 1, nt = wid & 1;   // 8 waves -> 4x2 tiles of 64x32
    const int colN = hbase + nt * 16 + l16;
    const int arow = mt * 16 + l16;
    int rowM[8];
#pragma unroll
    for (int r = 0; r < 8; ++r) rowM[r] = mt * 16 + r + 8 * half;

    float bh[NLAYERS];
#pragma unroll
    for (int j = 0; j < NLAYERS; ++j) bh[j] = b_h[j * HID + colN];

    U2 basekey; basekey.x = 0u; basekey.y = 1234u;   // jax.random.key(1234)
    v8f v = {};                                       // membrane state (f32)

    for (int t = 0; t < SEQ; ++t) {
        float wu[8];
#pragma unroll
        for (int r = 0; r < 8; ++r)
            wu[r] = __builtin_nontemporal_load(
                &WU[(t * BATCH + rowM[r]) * HID + colN]);
        if (t + 1 < SEQ)
            __builtin_prefetch(&WU[((t + 1) * BATCH + rowM[0]) * HID + colN], 0, 0);

#pragma unroll 1
        for (int j = 0; j < NLAYERS; ++j) {
            const int phase = t * NLAYERS + j;
            const unsigned short* frin  = frbuf + (phase & 1) * BATCH * HID;
            unsigned short*       frout = frbuf + ((phase & 1) ^ 1) * BATCH * HID;

            // ---- wh = fr @ W_h[j]^T for this wave's 16x16 tile -------------
            v8f c = {};
#pragma unroll 4
            for (int kb = 0; kb < HID / 32; ++kb) {
                Frag a, b;
                const u32x4* ap = (const u32x4*)(frin + arow * HID +
                                                 kb * 32 + 8 * half);
                a.q[0] = ap[0];   // K = kb*32 + 8*half + (0..7)
                a.q[1] = ap[2];   // K = kb*32 + 16 + 8*half + (0..7)
                const u32x4* bp = (const u32x4*)(sWh +
                    (j * 32 + nt * 16 + l16) * HID + kb * 32 + 8 * half);
                b.q[0] = bp[0];   // K = kb*32 + 8*half + (0..7)
                b.q[1] = bp[2];   // K = kb*32 + 16 + 8*half + (0..7)
                c = wmma_bf16(a, b, c);
            }

            // ---- leaky-integrate + threefry noise + relu -------------------
            U2 kpre  = jax_fold_in(basekey, (unsigned)(t * 2 * NLAYERS + 2 * j));
            U2 kpost = jax_fold_in(basekey, (unsigned)(t * 2 * NLAYERS + 2 * j + 1));
            float frv[8];
#pragma unroll
            for (int r = 0; r < 8; ++r) {
                unsigned idx = (unsigned)(rowM[r] * HID + colN);
                float wh = c[r] + bh[j];
                float nv = (1.0f - ALPHA) * v[r] + ALPHA * (wh + wu[r]);
                nv += ALPHA * PRE_NS * jax_normal(kpre, idx);
                v[r] = nv;
                frv[r] = fmaxf(nv, 0.0f) + POST_NS * jax_normal(kpost, idx);
            }

            // ---- publish fr slice (bf16) for the cluster; emit output ------
#pragma unroll
            for (int r = 0; r < 8; ++r)
                frout[rowM[r] * HID + colN] = f2bf(frv[r]);
            if (j == NLAYERS - 1) {
#pragma unroll
                for (int r = 0; r < 8; ++r)
                    __builtin_nontemporal_store(
                        frv[r], &out[(t * BATCH + rowM[r]) * HID + colN]);
            }
            cluster_sync();   // fr(phase+1) visible cluster-wide
        }
    }
}

// ===========================================================================
extern "C" void kernel_launch(void* const* d_in, const int* in_sizes, int n_in,
                              void* d_out, int out_size, void* d_ws,
                              size_t ws_size, hipStream_t stream) {
    const float* inp  = (const float*)d_in[0];   // (512, 64, 128)
    const float* W_in = (const float*)d_in[1];   // (512, 128)
    const float* b_in = (const float*)d_in[2];   // (512,)
    const float* W_h  = (const float*)d_in[3];   // (3, 512, 512)
    const float* b_h  = (const float*)d_in[4];   // (3, 512)
    float* out = (float*)d_out;                  // (512, 64, 512)

    float* WU = (float*)d_ws;                                       // 64 MB
    unsigned short* frbuf =
        (unsigned short*)((char*)d_ws + (size_t)SEQ * BATCH * HID * 4);  // 128 KB

    // Input projection: 65536 16x16 tiles, 8 waves/block
    lifrnn_inproj<<<(SEQ * BATCH / 16) * (HID / 16) / 8, 256, 0, stream>>>(
        inp, W_in, b_in, WU);

    // Persistent recurrent scan: 16 WGPs (cluster-sized), 96 KB LDS each
    const size_t lds = (size_t)NLAYERS * 32 * HID * sizeof(unsigned short);
    lifrnn_scan<<<NBLK_RNN, 256, lds, stream>>>(W_h, b_h, WU, frbuf, out);
}
